// SwinMixers_10591389352203
// MI455X (gfx1250) — matrix operand order repaired and checked by reading, not compiled, not run on hardware
//
#include <hip/hip_runtime.h>
#include <stdint.h>

// ---------------- problem constants (match reference) ----------------
#define T_TOTAL   32768
#define DMODEL    256
#define DINNER    512
#define LPATCH    1024
#define NWIN      32
#define NSTATE    16
#define DTRANK    16
#define XDBL      48
#define LN_EPS    1e-5f

// ---------------- vector types for WMMA ----------------
typedef __attribute__((ext_vector_type(16))) __bf16       v16bf;
typedef __attribute__((ext_vector_type(8)))  float        v8f;
typedef __attribute__((ext_vector_type(4)))  unsigned int u32x4;
typedef __attribute__((ext_vector_type(8)))  unsigned int u32x8;

__device__ __forceinline__ unsigned short f2bf(float f) {
  unsigned int u = __builtin_bit_cast(unsigned int, f);
  unsigned int r = u + 0x7FFFu + ((u >> 16) & 1u);   // round-to-nearest-even
  return (unsigned short)(r >> 16);
}

__device__ __forceinline__ float wave_sum(float x) {
#pragma unroll
  for (int off = 16; off >= 1; off >>= 1) x += __shfl_xor(x, off, 32);
  return x;
}

// A fragment: 16x32 bf16, row-major activations, stride ldk (elements).
// lane: m = lane&15 ; k-base = (lane>=16 ? 16 : 0); 16 consecutive k per lane
__device__ __forceinline__ v16bf load_a_frag(const unsigned short* A, int row0, int k0, int ldk) {
  int lane = threadIdx.x & 31;
  const unsigned short* p = A + (size_t)(row0 + (lane & 15)) * ldk + k0 + ((lane >> 4) << 4);
  u32x4 lo = *(const u32x4*)p;          // k .. k+7
  u32x4 hi = *(const u32x4*)(p + 8);    // k+8 .. k+15
  u32x8 t;
  t[0]=lo[0]; t[1]=lo[1]; t[2]=lo[2]; t[3]=lo[3];
  t[4]=hi[0]; t[5]=hi[1]; t[6]=hi[2]; t[7]=hi[3];
  return __builtin_bit_cast(v16bf, t);
}

// B fragment: 32x16 bf16 from pre-transposed (N-major) weights, stride ldk.
// lane: n = lane&15 ; k-base = (lane>=16 ? 16 : 0) ; 16 consecutive k per lane
__device__ __forceinline__ v16bf load_b_frag(const unsigned short* Bt, int col0, int k0, int ldk) {
  int lane = threadIdx.x & 31;
  const unsigned short* p = Bt + (size_t)(col0 + (lane & 15)) * ldk + k0 + ((lane >> 4) << 4);
  u32x8 t = *(const u32x8*)p;
  return __builtin_bit_cast(v16bf, t);
}

__device__ __forceinline__ v8f wmma_bf16(v16bf a, v16bf b, v8f c) {
  return __builtin_amdgcn_wmma_f32_16x16x32_bf16(false, a, false, b, (short)0, c, false, false);
}

// ---------------- weight convert: W[K][N] f32 -> Wt[N][K] bf16 ----------------
__global__ void k_convert_wt(const float* __restrict__ W, unsigned short* __restrict__ Wt,
                             int K, int N) {
  int idx = blockIdx.x * blockDim.x + threadIdx.x;
  if (idx >= K * N) return;
  int n = idx / K, k = idx - n * K;
  Wt[idx] = f2bf(W[(size_t)k * N + n]);
}

// ---------------- layernorm rows (gather) -> bf16 ----------------
__global__ void k_ln_gather_bf16(const float* __restrict__ X, const int* __restrict__ perm,
                                 const float* __restrict__ w, const float* __restrict__ b,
                                 unsigned short* __restrict__ out, int T) {
  int lane = threadIdx.x & 31;
  int row = blockIdx.x * (blockDim.x >> 5) + (threadIdx.x >> 5);
  if (row >= T) return;
  int src = perm ? perm[row] : row;
  const float* xr = X + (size_t)src * DMODEL;
  float v[8], s = 0.f, s2 = 0.f;
#pragma unroll
  for (int i = 0; i < 8; i++) { v[i] = xr[lane + i * 32]; s += v[i]; s2 += v[i] * v[i]; }
  s = wave_sum(s); s2 = wave_sum(s2);
  float mu = s * (1.f / DMODEL);
  float inv = rsqrtf(s2 * (1.f / DMODEL) - mu * mu + LN_EPS);
#pragma unroll
  for (int i = 0; i < 8; i++) {
    int c = lane + i * 32;
    out[(size_t)row * DMODEL + c] = f2bf((v[i] - mu) * inv * w[c] + b[c]);
  }
}

// ---------------- layernorm rows (scatter) -> bf16 ----------------
__global__ void k_ln_scatter_bf16(const float* __restrict__ X, const int* __restrict__ perm,
                                  const float* __restrict__ w, const float* __restrict__ b,
                                  unsigned short* __restrict__ out, int T) {
  int lane = threadIdx.x & 31;
  int row = blockIdx.x * (blockDim.x >> 5) + (threadIdx.x >> 5);
  if (row >= T) return;
  int dst = perm ? perm[row] : row;
  const float* xr = X + (size_t)row * DMODEL;
  float v[8], s = 0.f, s2 = 0.f;
#pragma unroll
  for (int i = 0; i < 8; i++) { v[i] = xr[lane + i * 32]; s += v[i]; s2 += v[i] * v[i]; }
  s = wave_sum(s); s2 = wave_sum(s2);
  float mu = s * (1.f / DMODEL);
  float inv = rsqrtf(s2 * (1.f / DMODEL) - mu * mu + LN_EPS);
#pragma unroll
  for (int i = 0; i < 8; i++) {
    int c = lane + i * 32;
    out[(size_t)dst * DMODEL + c] = f2bf((v[i] - mu) * inv * w[c] + b[c]);
  }
}

// ---------------- final layernorm -> f32 output ----------------
__global__ void k_ln_out(const float* __restrict__ X, const float* __restrict__ w,
                         const float* __restrict__ b, float* __restrict__ out, int T) {
  int lane = threadIdx.x & 31;
  int row = blockIdx.x * (blockDim.x >> 5) + (threadIdx.x >> 5);
  if (row >= T) return;
  const float* xr = X + (size_t)row * DMODEL;
  float v[8], s = 0.f, s2 = 0.f;
#pragma unroll
  for (int i = 0; i < 8; i++) { v[i] = xr[lane + i * 32]; s += v[i]; s2 += v[i] * v[i]; }
  s = wave_sum(s); s2 = wave_sum(s2);
  float mu = s * (1.f / DMODEL);
  float inv = rsqrtf(s2 * (1.f / DMODEL) - mu * mu + LN_EPS);
#pragma unroll
  for (int i = 0; i < 8; i++) {
    int c = lane + i * 32;
    out[(size_t)row * DMODEL + c] = (v[i] - mu) * inv * w[c] + b[c];
  }
}

// ---------------- WMMA GEMM: in_proj (Tx256 @ 256x1024), split xx / silu(z) ----------------
__global__ void k_gemm_inproj(const unsigned short* __restrict__ A,
                              const unsigned short* __restrict__ Bt,
                              float* __restrict__ xx, float* __restrict__ zs, int T) {
  int gw = blockIdx.x * (blockDim.x >> 5) + (threadIdx.x >> 5);
  const int tilesN = 1024 / 64;
  int tm = gw / tilesN, tn = gw - tm * tilesN;
  if (tm * 16 >= T) return;
  v8f acc[4] = {};
#pragma unroll
  for (int k0 = 0; k0 < 256; k0 += 32) {
    v16bf a = load_a_frag(A, tm * 16, k0, 256);
#pragma unroll
    for (int s = 0; s < 4; s++)
      acc[s] = wmma_bf16(a, load_b_frag(Bt, tn * 64 + s * 16, k0, 256), acc[s]);
  }
  int lane = threadIdx.x & 31;
  int n = lane & 15, mb = (lane >> 4) * 8;
#pragma unroll
  for (int s = 0; s < 4; s++) {
    int col = tn * 64 + s * 16 + n;
#pragma unroll
    for (int r = 0; r < 8; r++) {
      int row = tm * 16 + mb + r;
      float v = acc[s][r];
      if (col < DINNER) xx[(size_t)row * DINNER + col] = v;
      else              zs[(size_t)row * DINNER + (col - DINNER)] = v / (1.f + __expf(-v));
    }
  }
}

// ---------------- causal depthwise conv (K=4) + silu ----------------
__global__ void k_conv_silu(const float* __restrict__ xx, const float* __restrict__ cw,
                            const float* __restrict__ cb, float* __restrict__ xa,
                            unsigned short* __restrict__ xab, int T) {
  int idx = blockIdx.x * blockDim.x + threadIdx.x;
  if (idx >= T * DINNER) return;
  int d = idx & (DINNER - 1);
  int t = idx >> 9;
  int l = t & (LPATCH - 1);
  float acc = cb[d];
#pragma unroll
  for (int j = 0; j < 4; j++) {
    int dl = 3 - j;
    if (l >= dl) acc += cw[d * 4 + j] * xx[(size_t)(t - dl) * DINNER + d];
  }
  float s = acc / (1.f + __expf(-acc));
  xa[idx] = s;
  xab[idx] = f2bf(s);
}

// ---------------- WMMA GEMM: x_proj (Tx512 @ 512x48) ----------------
__global__ void k_gemm_xproj(const unsigned short* __restrict__ A,
                             const unsigned short* __restrict__ Bt,
                             float* __restrict__ xdbl, int T) {
  int tm = blockIdx.x * (blockDim.x >> 5) + (threadIdx.x >> 5);
  if (tm * 16 >= T) return;
  v8f acc[3] = {};
#pragma unroll
  for (int k0 = 0; k0 < 512; k0 += 32) {
    v16bf a = load_a_frag(A, tm * 16, k0, 512);
#pragma unroll
    for (int s = 0; s < 3; s++)
      acc[s] = wmma_bf16(a, load_b_frag(Bt, s * 16, k0, 512), acc[s]);
  }
  int lane = threadIdx.x & 31;
  int n = lane & 15, mb = (lane >> 4) * 8;
#pragma unroll
  for (int s = 0; s < 3; s++) {
#pragma unroll
    for (int r = 0; r < 8; r++)
      xdbl[(size_t)(tm * 16 + mb + r) * XDBL + s * 16 + n] = acc[s][r];
  }
}

// ---------------- dt = softplus(dt_raw @ dt_w + dt_b) ----------------
__global__ void k_dt(const float* __restrict__ xdbl, const float* __restrict__ dtw,
                     const float* __restrict__ dtb, float* __restrict__ dt, int T) {
  int idx = blockIdx.x * blockDim.x + threadIdx.x;
  if (idx >= T * DINNER) return;
  int d = idx & (DINNER - 1);
  int t = idx >> 9;
  float acc = dtb[d];
  const float* xr = xdbl + (size_t)t * XDBL;
#pragma unroll
  for (int r = 0; r < DTRANK; r++) acc += xr[r] * dtw[r * DINNER + d];
  dt[idx] = (acc > 20.f) ? acc : log1pf(__expf(acc));
}

// ---------------- selective scan + skip(D) + silu(z) gating -> bf16 ----------------
__global__ void __launch_bounds__(DINNER)
k_scan(const float* __restrict__ xa, const float* __restrict__ dt,
       const float* __restrict__ xdbl, const float* __restrict__ zs,
       const float* __restrict__ Alog, const float* __restrict__ Dp,
       unsigned short* __restrict__ yg) {
  __shared__ float sB[NSTATE];
  __shared__ float sC[NSTATE];
  int nwin = blockIdx.x;          // 0..31
  int d = threadIdx.x;            // 0..511
  float A[NSTATE], h[NSTATE];
#pragma unroll
  for (int k = 0; k < NSTATE; k++) { A[k] = -__expf(Alog[d * NSTATE + k]); h[k] = 0.f; }
  float Dv = Dp[d];
  for (int l = 0; l < LPATCH; l++) {
    int t = nwin * LPATCH + l;
    if (threadIdx.x < 32) {
      const float* xr = xdbl + (size_t)t * XDBL;
      if (threadIdx.x < 16) sB[threadIdx.x] = xr[DTRANK + threadIdx.x];
      else                  sC[threadIdx.x - 16] = xr[DTRANK + NSTATE + threadIdx.x - 16];
    }
    __syncthreads();
    float u   = xa[(size_t)t * DINNER + d];
    float dtv = dt[(size_t)t * DINNER + d];
    float du  = dtv * u;
    float y = 0.f;
#pragma unroll
    for (int k = 0; k < NSTATE; k++) {
      float dA = __expf(dtv * A[k]);
      h[k] = dA * h[k] + du * sB[k];
      y += h[k] * sC[k];
    }
    y += u * Dv;
    y *= zs[(size_t)t * DINNER + d];
    yg[(size_t)t * DINNER + d] = f2bf(y);
    __syncthreads();
  }
}

// ---------------- WMMA GEMM: out_proj (Tx512 @ 512x256) + residual ----------------
__global__ void k_gemm_out(const unsigned short* __restrict__ A,
                           const unsigned short* __restrict__ Bt,
                           const float* __restrict__ hidden, const int* __restrict__ perm,
                           float* __restrict__ hpre, int T) {
  int gw = blockIdx.x * (blockDim.x >> 5) + (threadIdx.x >> 5);
  const int tilesN = 256 / 64;
  int tm = gw / tilesN, tn = gw - tm * tilesN;
  if (tm * 16 >= T) return;
  v8f acc[4] = {};
#pragma unroll
  for (int k0 = 0; k0 < 512; k0 += 32) {
    v16bf a = load_a_frag(A, tm * 16, k0, 512);
#pragma unroll
    for (int s = 0; s < 4; s++)
      acc[s] = wmma_bf16(a, load_b_frag(Bt, tn * 64 + s * 16, k0, 512), acc[s]);
  }
  int lane = threadIdx.x & 31;
  int n = lane & 15, mb = (lane >> 4) * 8;
#pragma unroll
  for (int s = 0; s < 4; s++) {
    int col = tn * 64 + s * 16 + n;
#pragma unroll
    for (int r = 0; r < 8; r++) {
      int row = tm * 16 + mb + r;
      int src = perm ? perm[row] : row;
      hpre[(size_t)row * DMODEL + col] = acc[s][r] + hidden[(size_t)src * DMODEL + col];
    }
  }
}

// ---------------- WMMA GEMM: merge ([h0|h1] Tx512 @ 512x256) + bias ----------------
__global__ void k_gemm_merge(const unsigned short* __restrict__ A0,
                             const unsigned short* __restrict__ A1,
                             const unsigned short* __restrict__ Bt,
                             const float* __restrict__ mbias,
                             float* __restrict__ hpre, int T) {
  int gw = blockIdx.x * (blockDim.x >> 5) + (threadIdx.x >> 5);
  const int tilesN = 256 / 64;
  int tm = gw / tilesN, tn = gw - tm * tilesN;
  if (tm * 16 >= T) return;
  v8f acc[4] = {};
#pragma unroll
  for (int k0 = 0; k0 < 512; k0 += 32) {
    const unsigned short* Asrc = (k0 < 256) ? A0 : A1;
    int kk = (k0 < 256) ? k0 : (k0 - 256);
    v16bf a = load_a_frag(Asrc, tm * 16, kk, 256);
#pragma unroll
    for (int s = 0; s < 4; s++)
      acc[s] = wmma_bf16(a, load_b_frag(Bt, tn * 64 + s * 16, k0, 512), acc[s]);
  }
  int lane = threadIdx.x & 31;
  int n = lane & 15, mb = (lane >> 4) * 8;
#pragma unroll
  for (int s = 0; s < 4; s++) {
    int col = tn * 64 + s * 16 + n;
#pragma unroll
    for (int r = 0; r < 8; r++)
      hpre[(size_t)(tm * 16 + mb + r) * DMODEL + col] = acc[s][r] + mbias[col];
  }
}

// =====================================================================
extern "C" void kernel_launch(void* const* d_in, const int* in_sizes, int n_in,
                              void* d_out, int out_size, void* d_ws, size_t ws_size,
                              hipStream_t stream) {
  (void)in_sizes; (void)out_size; (void)ws_size;
  if (n_in < 34) return;

  const float* hidden = (const float*)d_in[0];
  const int*   shift  = (const int*)d_in[1];
  // d_in[2] = shift_back (unused: scatter via shift), d_in[3] = patch_size

  // per-block params: base 4 (b0), 15 (b1); order:
  // norm_w, norm_b, in_proj, conv_w, conv_b, x_proj, dt_w, dt_b, A_log, D, out_proj
  const float* P[2][11];
  for (int b = 0; b < 2; b++)
    for (int i = 0; i < 11; i++) P[b][i] = (const float*)d_in[4 + b * 11 + i];
  const float* nf_w[2] = { (const float*)d_in[26], (const float*)d_in[28] };
  const float* nf_b[2] = { (const float*)d_in[27], (const float*)d_in[29] };
  const float* merge_w = (const float*)d_in[30];
  const float* merge_b = (const float*)d_in[31];
  const float* nff_w   = (const float*)d_in[32];
  const float* nff_b   = (const float*)d_in[33];

  // ---------------- workspace layout ----------------
  char* wp = (char*)d_ws;
  auto alloc = [&](size_t bytes) -> void* {
    void* r = (void*)wp;
    wp += (bytes + 255) & ~(size_t)255;
    return r;
  };
  const size_t T = T_TOTAL;
  unsigned short* wt_in[2]  = { (unsigned short*)alloc(1024 * 256 * 2),
                                (unsigned short*)alloc(1024 * 256 * 2) };
  unsigned short* wt_xp[2]  = { (unsigned short*)alloc(XDBL * 512 * 2),
                                (unsigned short*)alloc(XDBL * 512 * 2) };
  unsigned short* wt_out[2] = { (unsigned short*)alloc(256 * 512 * 2),
                                (unsigned short*)alloc(256 * 512 * 2) };
  unsigned short* wt_mrg    = (unsigned short*)alloc(256 * 512 * 2);
  unsigned short* hb[2]     = { (unsigned short*)alloc(T * DMODEL * 2),
                                (unsigned short*)alloc(T * DMODEL * 2) };
  unsigned short* xn   = (unsigned short*)alloc(T * DMODEL * 2);
  float*          xx   = (float*)alloc(T * DINNER * 4);
  float*          zsil = (float*)alloc(T * DINNER * 4);
  float*          xa   = (float*)alloc(T * DINNER * 4);
  unsigned short* xab  = (unsigned short*)alloc(T * DINNER * 2);
  float*          xdbl = (float*)alloc(T * XDBL * 4);
  float*          dtb  = (float*)alloc(T * DINNER * 4);
  unsigned short* yg   = (unsigned short*)alloc(T * DINNER * 2);
  float*          hpre = (float*)alloc(T * DMODEL * 4);

  // ---------------- weight conversion (f32 -> bf16, transposed N-major) ----------------
  for (int b = 0; b < 2; b++) {
    k_convert_wt<<<(256 * 1024 + 255) / 256, 256, 0, stream>>>(P[b][2], wt_in[b], 256, 1024);
    k_convert_wt<<<(512 * XDBL + 255) / 256, 256, 0, stream>>>(P[b][5], wt_xp[b], 512, XDBL);
    k_convert_wt<<<(512 * 256 + 255) / 256, 256, 0, stream>>>(P[b][10], wt_out[b], 512, 256);
  }
  k_convert_wt<<<(512 * 256 + 255) / 256, 256, 0, stream>>>(merge_w, wt_mrg, 512, 256);

  const int ROWS_BLK = (int)T / 8;                 // 8 rows per 256-thread block
  const int IN_BLKS  = ((int)T / 16) * 16 / 8;     // (T/16 tilesM)*(16 tilesN)/8 waves
  const int XP_BLKS  = ((int)T / 16) / 8;
  const int ON_BLKS  = ((int)T / 16) * 4 / 8;

  // ---------------- two mamba branches ----------------
  for (int b = 0; b < 2; b++) {
    const int* perm = (b == 0) ? (const int*)nullptr : shift;
    k_ln_gather_bf16<<<ROWS_BLK, 256, 0, stream>>>(hidden, perm, P[b][0], P[b][1], xn, (int)T);
    k_gemm_inproj<<<IN_BLKS, 256, 0, stream>>>(xn, wt_in[b], xx, zsil, (int)T);
    k_conv_silu<<<(int)(T * DINNER / 256), 256, 0, stream>>>(xx, P[b][3], P[b][4], xa, xab, (int)T);
    k_gemm_xproj<<<XP_BLKS, 256, 0, stream>>>(xab, wt_xp[b], xdbl, (int)T);
    k_dt<<<(int)(T * DINNER / 256), 256, 0, stream>>>(xdbl, P[b][6], P[b][7], dtb, (int)T);
    k_scan<<<NWIN, DINNER, 0, stream>>>(xa, dtb, xdbl, zsil, P[b][8], P[b][9], yg);
    k_gemm_out<<<ON_BLKS, 256, 0, stream>>>(yg, wt_out[b], hidden, perm, hpre, (int)T);
    k_ln_scatter_bf16<<<ROWS_BLK, 256, 0, stream>>>(hpre, perm, nf_w[b], nf_b[b], hb[b], (int)T);
  }

  // ---------------- merge + final layernorm ----------------
  k_gemm_merge<<<ON_BLKS, 256, 0, stream>>>(hb[0], hb[1], wt_mrg, merge_b, hpre, (int)T);
  k_ln_out<<<ROWS_BLK, 256, 0, stream>>>(hpre, nff_w, nff_b, (float*)d_out, (int)T);
}
